// Conv_block_44976897524621
// MI455X (gfx1250) — matrix operand order repaired and checked by reading, not compiled, not run on hardware
//
#include <hip/hip_runtime.h>
#include <hip/hip_bf16.h>

// ---------------------------------------------------------------------------
// Routed conv (high: 3x3 64->64, low: 3x3 64->16 then 1x1 16->64), per-pixel
// selection.  Implicit-GEMM with fused N=80 (64 high + 16 low-mid) bf16 WMMA,
// f32 accumulate, on gfx1250 wave32.
//
// K ordering: k = pos*64 + ci  (pos = kh*3+kw).  Each K-chunk of 32 covers a
// single spatial tap and 32 consecutive channels, so with a channel-innermost
// LDS tile every lane's A fragment is two 16B-contiguous runs -> 2x ds_load_b128.
// B fragments are pre-packed in per-lane WMMA layout and double-buffered one
// K-step ahead so WMMAs overlap the global/LDS operand fetch.
// ---------------------------------------------------------------------------

#define BATCH 4
#define CIN   64
#define HH    384
#define WW    384
#define COUT  64
#define CL    16              // low bottleneck channels
#define KTOT  576             // CIN * 3 * 3
#define KSTEPS 18             // KTOT / 32
#define TW    128             // pixels per workgroup (8 waves * 16)
#define NPIX  (BATCH * HH * WW)
#define CSTRIDE 72            // padded channel stride (16B-aligned, bank-spread)

// workspace layout (bytes)
#define BPACK_BYTES  (KSTEPS * 5 * 32 * 16 * 2)   // 92160
#define B2PACK_OFF   BPACK_BYTES                  // 92160, 32B aligned
#define B2PACK_BYTES (4 * 32 * 16 * 2)            // 4096
#define FLAGS_OFF    (B2PACK_OFF + B2PACK_BYTES)  // 96256

typedef __attribute__((ext_vector_type(16))) __bf16 v16bf;
typedef __attribute__((ext_vector_type(8)))  __bf16 v8bf;
typedef __attribute__((ext_vector_type(8)))  float  v8f;

// ---------------------------------------------------------------------------
// Prep: zero routing flags; pack high_w/low1_w into main-GEMM B fragments and
// low2_w into the 1x1 B fragments, in per-lane WMMA B layout:
//   frag f = (kstep*5 + nt), lane L holds 16 bf16 = B[k][n] with
//   n = nt*16 + (L&15), k = kstep*32 + ((L&16)?8:0) + (pos16&7) + ((pos16&8)?16:0), pos16 = 0..15.
//   Global k decodes as: spatial = k>>6, ci = k&63.
// ---------------------------------------------------------------------------
__global__ void init_pack_kernel(const float* __restrict__ high_w,
                                 const float* __restrict__ low1_w,
                                 const float* __restrict__ low2_w,
                                 unsigned char* __restrict__ flags,
                                 __bf16* __restrict__ Bpack,
                                 __bf16* __restrict__ B2pack) {
    int gid = blockIdx.x * blockDim.x + threadIdx.x;
    if (gid < NPIX) flags[gid] = 0;

    if (gid < KSTEPS * 5 * 32 * 16) {
        int pos16 = gid & 15;
        int lane  = (gid >> 4) & 31;
        int fi    = gid >> 9;            // kstep*5 + nt
        int nt    = fi % 5;
        int kstep = fi / 5;
        int n = nt * 16 + (lane & 15);
        int k = kstep * 32 + ((lane & 16) ? 8 : 0) + (pos16 & 7) + ((pos16 & 8) << 1);
        int sp = k >> 6;                 // spatial tap kh*3+kw
        int ci = k & 63;
        // weights are OIHW: flat [n][ci][kh][kw] = n*576 + ci*9 + sp
        float v = (n < COUT) ? high_w[n * KTOT + ci * 9 + sp]
                             : low1_w[(n - COUT) * KTOT + ci * 9 + sp];
        Bpack[gid] = (__bf16)v;
    }
    if (gid < 4 * 32 * 16) {
        int pos16 = gid & 15;
        int lane  = (gid >> 4) & 31;
        int nt    = gid >> 9;
        int kk    = ((lane & 16) ? 8 : 0) + (pos16 & 7) + ((pos16 & 8) << 1);   // K padded 16 -> 32
        int n     = nt * 16 + (lane & 15);
        float v   = (kk < CL) ? low2_w[n * CL + kk] : 0.0f;
        B2pack[gid] = (__bf16)v;
    }
}

__global__ void scatter_flags_kernel(const int* __restrict__ mask_idx,
                                     unsigned char* __restrict__ flags, int n) {
    int gid = blockIdx.x * blockDim.x + threadIdx.x;
    if (gid < n) flags[mask_idx[gid]] = 1;
}

// ---------------------------------------------------------------------------
// Main fused kernel.  grid = (WW/TW, HH, BATCH), block = 256 (8 wave32).
// ---------------------------------------------------------------------------
__global__ __launch_bounds__(256)
void routed_conv_kernel(const float* __restrict__ x,
                        const unsigned char* __restrict__ flags,
                        const __bf16* __restrict__ Bpack,
                        const __bf16* __restrict__ B2pack,
                        float* __restrict__ out) {
    // halo tile, channel-innermost: xs[(r*(TW+2)+c)*CSTRIDE + ci]
    __shared__ alignas(16) __bf16 xs[3 * (TW + 2) * CSTRIDE];   // ~56.2 KB
    __shared__ float midS[8][16 * 17];                          // mid reshape
    __shared__ alignas(16) float outS[8][COUT * 20];            // store staging

    const int tid = threadIdx.x;
    const int b  = blockIdx.z;
    const int h  = blockIdx.y;
    const int w0 = blockIdx.x * TW;

    // ---- stage x halo tile (zero-padded SAME borders) as bf16 into LDS ----
    // thread order: (ci, r, c) with c fastest -> coalesced global reads
    for (int e = tid; e < CIN * 3 * (TW + 2); e += 256) {
        int ci  = e / (3 * (TW + 2));
        int rem = e - ci * (3 * (TW + 2));
        int r   = rem / (TW + 2);
        int c   = rem - r * (TW + 2);
        int gh  = h - 1 + r;
        int gw  = w0 - 1 + c;
        float v = 0.0f;
        if ((unsigned)gh < (unsigned)HH && (unsigned)gw < (unsigned)WW)
            v = x[((b * CIN + ci) * HH + gh) * WW + gw];
        xs[(r * (TW + 2) + c) * CSTRIDE + ci] = (__bf16)v;
    }
    __syncthreads();

    const int wave  = tid >> 5;
    const int lane  = tid & 31;
    const int col16 = lane & 15;                 // A row / C column sub-index
    const int Mbase = wave * 16;
    const int lw    = Mbase + col16;             // local w of this lane's A row
    const int ci_lo = (lane & 16) ? 8 : 0;       // per-lane K run offset (A layout)

    const v16bf* __restrict__ Bp  = (const v16bf*)Bpack;
    const v16bf* __restrict__ B2p = (const v16bf*)B2pack;

    // A-fragment fetch: lane's two contiguous 8-channel runs in LDS
    auto loadA = [&](int kstep) -> v16bf {
        const int sp     = kstep >> 1;           // spatial tap
        const int kh     = sp / 3;
        const int kw     = sp - kh * 3;
        const int cibase = (kstep & 1) << 5;     // 0 or 32
        const int base = (kh * (TW + 2) + lw + kw) * CSTRIDE + cibase + ci_lo;
        v8bf alo = *(const v8bf*)&xs[base];
        v8bf ahi = *(const v8bf*)&xs[base + 16];
        return __builtin_shufflevector(alo, ahi,
            0, 1, 2, 3, 4, 5, 6, 7, 8, 9, 10, 11, 12, 13, 14, 15);
    };

    const v8f zacc = {0.f, 0.f, 0.f, 0.f, 0.f, 0.f, 0.f, 0.f};
    v8f acc[5];
#pragma unroll
    for (int i = 0; i < 5; ++i) acc[i] = zacc;

    // ---- main K loop: 18 x (K=32) bf16 WMMA, fused N = 64 high + 16 low ----
    // Double-buffered operands: fetch step k+1 while MMAs of step k execute.
    v16bf bcur[5];
#pragma unroll
    for (int nt = 0; nt < 5; ++nt) bcur[nt] = Bp[nt * 32 + lane];
    v16bf acur = loadA(0);

#pragma unroll
    for (int kstep = 0; kstep < KSTEPS; ++kstep) {
        const int kn = (kstep + 1 < KSTEPS) ? kstep + 1 : kstep;
        v16bf bnxt[5];
#pragma unroll
        for (int nt = 0; nt < 5; ++nt) bnxt[nt] = Bp[(kn * 5 + nt) * 32 + lane];
        v16bf anxt = loadA(kn);

#pragma unroll
        for (int nt = 0; nt < 5; ++nt) {
            acc[nt] = __builtin_amdgcn_wmma_f32_16x16x32_bf16(
                false, acur, false, bcur[nt], (short)0, acc[nt], false, false);
        }
        acur = anxt;
#pragma unroll
        for (int nt = 0; nt < 5; ++nt) bcur[nt] = bnxt[nt];
    }

    // ---- epilogue: low path 1x1 (16->64) via WMMA on the mid tile ----------
    // C layout: lane holds channel (lane&15) for rows m = r + 8*(lane>=16).
    float* ms = midS[wave];
#pragma unroll
    for (int r = 0; r < 8; ++r) {
        int m = r + ((lane >> 4) << 3);
        ms[m * 17 + col16] = acc[4][r];          // [pixel][midchan], padded 17
    }
    // rebuild as A fragment (M=16 pixels, K=16 real channels, padded to 32)
    const int kb = (lane & 16) ? 8 : 0;
    v16bf ma;
#pragma unroll
    for (int v = 0; v < 8; ++v) {
#pragma unroll
        for (int j = 0; j < 2; ++j) {
            int kk = kb + ((v & 4) << 2) + ((v & 3) << 1) + j;
            float mv = (kk < CL) ? ms[col16 * 17 + kk] : 0.0f;
            ma[2 * v + j] = (__bf16)mv;
        }
    }
    v8f accL[4];
#pragma unroll
    for (int nt = 0; nt < 4; ++nt) {
        v16bf bfrag = B2p[nt * 32 + lane];
        accL[nt] = __builtin_amdgcn_wmma_f32_16x16x32_bf16(
            false, ma, false, bfrag, (short)0, zacc, false, false);
    }

    // ---- per-pixel routing + transpose through LDS for coalesced stores ----
    const int base_pix = (b * HH + h) * WW + w0 + Mbase;
    bool hi[8];
#pragma unroll
    for (int r = 0; r < 8; ++r) {
        int m = r + ((lane >> 4) << 3);
        hi[r] = flags[base_pix + m] != 0;
    }
    float* os = outS[wave];
#pragma unroll
    for (int nt = 0; nt < 4; ++nt) {
#pragma unroll
        for (int r = 0; r < 8; ++r) {
            int m  = r + ((lane >> 4) << 3);
            int co = nt * 16 + col16;
            os[co * 20 + m] = hi[r] ? acc[nt][r] : accL[nt][r];   // 20: bank pad
        }
    }
    // 8 channels x 16 w per iteration, 64B-contiguous float4 stores
#pragma unroll
    for (int i = 0; i < 8; ++i) {
        int co = i * 8 + (lane >> 2);
        int mw = (lane & 3) * 4;
        float4 vv = *(const float4*)&os[co * 20 + mw];
        *(float4*)&out[((b * COUT + co) * HH + h) * WW + w0 + Mbase + mw] = vv;
    }
}

// ---------------------------------------------------------------------------
extern "C" void kernel_launch(void* const* d_in, const int* in_sizes, int n_in,
                              void* d_out, int out_size, void* d_ws, size_t ws_size,
                              hipStream_t stream) {
    const float* x       = (const float*)d_in[0];
    const float* high_w  = (const float*)d_in[1];
    const float* low1_w  = (const float*)d_in[2];
    const float* low2_w  = (const float*)d_in[3];
    const int*   mask_ix = (const int*)d_in[4];
    // d_in[5] (inv_mask_idx) not needed: flags default to low path.

    char* ws = (char*)d_ws;
    __bf16* Bpack        = (__bf16*)ws;
    __bf16* B2pack       = (__bf16*)(ws + B2PACK_OFF);
    unsigned char* flags = (unsigned char*)(ws + FLAGS_OFF);

    init_pack_kernel<<<(NPIX + 255) / 256, 256, 0, stream>>>(
        high_w, low1_w, low2_w, flags, Bpack, B2pack);

    int nmask = in_sizes[4];
    scatter_flags_kernel<<<(nmask + 255) / 256, 256, 0, stream>>>(
        mask_ix, flags, nmask);

    dim3 grid(WW / TW, HH, BATCH);
    routed_conv_kernel<<<grid, 256, 0, stream>>>(
        x, flags, Bpack, B2pack, (float*)d_out);
}
